// GINE_85856396247548
// MI455X (gfx1250) — compile-verified, matrix operand order checked by reading, not run on hardware
//
#include <hip/hip_runtime.h>
#include <hip/hip_bf16.h>

#define N_NODES   100000
#define N_EDGES   625000
#define HID       128
#define N_LAYERS  4
#define NUM_GRAPHS 512

typedef __attribute__((ext_vector_type(16))) __bf16        v16bf;
typedef __attribute__((ext_vector_type(8)))  float         v8f;
typedef __attribute__((ext_vector_type(4)))  unsigned int  u32x4;

union Frag16 { v16bf v; u32x4 q[2]; };

__device__ __forceinline__ unsigned int f32_to_bf16(float f) {
    unsigned int u = __float_as_uint(f);
    unsigned int r = u + 0x7FFFu + ((u >> 16) & 1u);   // round-to-nearest-even
    return r >> 16;
}

// Fragment-ordered LDS layout: each (row r, K-block kb, half hw) owns one
// 32-byte granule holding the 8 dwords of that lane's WMMA fragment.
// XOR swizzle (bits 0..2 only) spreads rows across the 64 LDS banks.
// Key property: granule(r,kb) == granule(r,0) + kb*(2R)  (XOR never reaches
// the kb term's bits), so per-kb addresses are base + constant byte offset.
__device__ __forceinline__ int frag_granule(int r, int kb, int hw, int R) {
    return ((((kb * R) + r) << 1) | hw) ^ ((r >> 2) & 7);
}

// For even element K-offset k, the WMMA 16-bit A/B layout (ISA 7.12.2) puts
// the (k, k+1) bf16 pair in one dword: half hw=(k>>3)&1, dword v.
__device__ __forceinline__ void frag_slot_pair(int k, int& hw, int& v) {
    hw = (k >> 3) & 1;
    v  = ((k & 16) ? 4 : 0) + ((k & 7) >> 1);
}

// ---------------------------------------------------------------- embedding
__global__ __launch_bounds__(256) void embed_kernel(
    const int* __restrict__ x, const float* __restrict__ node_emb,
    float* __restrict__ h)
{
    int t = blockIdx.x * 256 + threadIdx.x;
    if (t >= N_NODES * HID) return;
    int node = t >> 7, c = t & 127;
    h[t] = node_emb[x[node] * HID + c];
}

// ---------------------------------------------------------------- edge scatter
__global__ __launch_bounds__(256) void scatter_add_kernel(
    const float* __restrict__ h, const int* __restrict__ src,
    const int* __restrict__ dst, float* __restrict__ agg)
{
    int t = blockIdx.x * 256 + threadIdx.x;
    if (t >= N_EDGES * HID) return;
    int e = t >> 7, c = t & 127;
    atomicAdd(&agg[dst[e] * HID + c], h[src[e] * HID + c]);   // L2-resident atomics
}

// ---------------------------------------------------------------- fused dual GEMM
// h_out = relu(agg @ W1 + b1) @ W2 + b2   (agg already holds z-input = sum + h)
__global__ __launch_bounds__(256) void gine_layer_gemm(
    const float* __restrict__ agg,
    const float* __restrict__ W1, const float* __restrict__ b1,
    const float* __restrict__ W2, const float* __restrict__ b2,
    float* __restrict__ h_out)
{
    __shared__ __align__(32) unsigned short sA[64 * HID];    // 16 KB, fragment order
    __shared__ __align__(32) unsigned short sW[HID * HID];   // 32 KB, fragment order
    __shared__ __align__(32) unsigned short sZ[64 * HID];    // 16 KB, fragment order

    const int tid  = threadIdx.x;
    const int lane = tid & 31;
    const int wave = tid >> 5;                 // 8 waves, wave == column tile
    const int row0 = blockIdx.x * 64;

    // ---- stage A tile: one float2 read -> one packed ds_store_b32
    for (int i = tid; i < 64 * 64; i += 256) {             // 4096 dword slots
        int m = i >> 6, k = (i & 63) << 1;
        int gr = row0 + m; if (gr >= N_NODES) gr = N_NODES - 1;
        float2 f2 = *(const float2*)&agg[gr * HID + k];
        unsigned int pk = f32_to_bf16(f2.x) | (f32_to_bf16(f2.y) << 16);
        int hwk, v; frag_slot_pair(k, hwk, v);
        *(unsigned int*)&sA[frag_granule(m, k >> 5, hwk, 64) * 16 + v * 2] = pk;
    }
    // ---- stage W1: two coalesced row reads -> one packed ds_store_b32
    for (int i = tid; i < 64 * HID; i += 256) {            // 8192 dword slots
        int k = (i >> 7) << 1, nn = i & 127;
        unsigned int pk = f32_to_bf16(W1[k * HID + nn]) |
                          (f32_to_bf16(W1[(k + 1) * HID + nn]) << 16);
        int hwk, v; frag_slot_pair(k, hwk, v);
        *(unsigned int*)&sW[frag_granule(nn, k >> 5, hwk, HID) * 16 + v * 2] = pk;
    }
    __syncthreads();

    const int nl = lane & 15;
    const int hw = lane >> 4;
    const int n  = wave * 16 + nl;             // output column for C/D layout

    // ---- precompute fragment base addresses once; kb offsets are constants
    const char* baseA[4];
    const char* baseZ[4];
    #pragma unroll
    for (int rt = 0; rt < 4; ++rt) {
        baseA[rt] = (const char*)sA + frag_granule(rt * 16 + nl, 0, hw, 64) * 32;
        baseZ[rt] = (const char*)sZ + frag_granule(rt * 16 + nl, 0, hw, 64) * 32;
    }
    const char* baseB = (const char*)sW + frag_granule(n, 0, hw, HID) * 32;

    v8f acc[4] = {};
    #pragma unroll
    for (int kb = 0; kb < 4; ++kb) {
        Frag16 bf, af[4];
        bf.q[0] = *(const u32x4*)(baseB + kb * 8192);
        bf.q[1] = *(const u32x4*)(baseB + kb * 8192 + 16);
        #pragma unroll
        for (int rt = 0; rt < 4; ++rt) {
            af[rt].q[0] = *(const u32x4*)(baseA[rt] + kb * 4096);
            af[rt].q[1] = *(const u32x4*)(baseA[rt] + kb * 4096 + 16);
        }
        #pragma unroll
        for (int rt = 0; rt < 4; ++rt)
            acc[rt] = __builtin_amdgcn_wmma_f32_16x16x32_bf16(
                false, af[rt].v, false, bf.v, (short)0, acc[rt], false, false);
    }
    __syncthreads();                           // all waves done reading sW/sA

    // ---- epilogue 1: bias + relu -> sZ (fragment order; Z's K index is n)
    {
        float bias1 = b1[n];
        int hwz = (n >> 3) & 1;
        int vz  = ((n & 16) ? 4 : 0) + ((n & 7) >> 1);
        int parz = n & 1;
        #pragma unroll
        for (int rt = 0; rt < 4; ++rt) {
            #pragma unroll
            for (int r = 0; r < 8; ++r) {
                int m = rt * 16 + r + 8 * hw;  // C/D layout: VGPR r -> M = r + 8*half
                float zv = acc[rt][r] + bias1;
                sZ[frag_granule(m, n >> 5, hwz, 64) * 16 + vz * 2 + parz] =
                    (unsigned short)f32_to_bf16(zv > 0.f ? zv : 0.f);
            }
        }
    }
    // ---- stage W2 into the same buffer
    for (int i = tid; i < 64 * HID; i += 256) {
        int k = (i >> 7) << 1, nn = i & 127;
        unsigned int pk = f32_to_bf16(W2[k * HID + nn]) |
                          (f32_to_bf16(W2[(k + 1) * HID + nn]) << 16);
        int hwk, v; frag_slot_pair(k, hwk, v);
        *(unsigned int*)&sW[frag_granule(nn, k >> 5, hwk, HID) * 16 + v * 2] = pk;
    }
    __syncthreads();

    v8f acc2[4] = {};
    #pragma unroll
    for (int kb = 0; kb < 4; ++kb) {
        Frag16 bf, af[4];
        bf.q[0] = *(const u32x4*)(baseB + kb * 8192);
        bf.q[1] = *(const u32x4*)(baseB + kb * 8192 + 16);
        #pragma unroll
        for (int rt = 0; rt < 4; ++rt) {
            af[rt].q[0] = *(const u32x4*)(baseZ[rt] + kb * 4096);
            af[rt].q[1] = *(const u32x4*)(baseZ[rt] + kb * 4096 + 16);
        }
        #pragma unroll
        for (int rt = 0; rt < 4; ++rt)
            acc2[rt] = __builtin_amdgcn_wmma_f32_16x16x32_bf16(
                false, af[rt].v, false, bf.v, (short)0, acc2[rt], false, false);
    }

    float bias2 = b2[n];
    #pragma unroll
    for (int rt = 0; rt < 4; ++rt) {
        #pragma unroll
        for (int r = 0; r < 8; ++r) {
            int m = rt * 16 + r + 8 * hw;
            int gr = row0 + m;
            if (gr < N_NODES) h_out[gr * HID + n] = acc2[rt][r] + bias2;
        }
    }
}

// ---------------------------------------------------------------- pooling
__global__ __launch_bounds__(256) void zero_kernel(float* __restrict__ p, int nelem) {
    int t = blockIdx.x * 256 + threadIdx.x;
    if (t < nelem) p[t] = 0.f;
}

__global__ __launch_bounds__(256) void pool_add_kernel(
    const float* __restrict__ h, const int* __restrict__ batch,
    float* __restrict__ g)
{
    int t = blockIdx.x * 256 + threadIdx.x;
    if (t >= N_NODES * HID) return;
    int node = t >> 7, c = t & 127;
    atomicAdd(&g[batch[node] * HID + c], h[t]);
}

// ---------------------------------------------------------------- MLP head (tiny)
__global__ __launch_bounds__(64) void mlp_head_kernel(
    const float* __restrict__ g,
    const float* __restrict__ W1, const float* __restrict__ b1,
    const float* __restrict__ W2, const float* __restrict__ b2,
    const float* __restrict__ W3, const float* __restrict__ b3,
    float* __restrict__ out)
{
    __shared__ float gin[128], z1[64], z2[32];
    int t = threadIdx.x, gi = blockIdx.x;
    gin[t]      = g[gi * HID + t];
    gin[t + 64] = g[gi * HID + t + 64];
    __syncthreads();
    float s = b1[t];
    for (int k = 0; k < 128; ++k) s += gin[k] * W1[k * 64 + t];
    z1[t] = s > 0.f ? s : 0.f;
    __syncthreads();
    if (t < 32) {
        float s2 = b2[t];
        for (int k = 0; k < 64; ++k) s2 += z1[k] * W2[k * 32 + t];
        z2[t] = s2 > 0.f ? s2 : 0.f;
    }
    __syncthreads();
    if (t == 0) {
        float s3 = b3[0];
        for (int k = 0; k < 32; ++k) s3 += z2[k] * W3[k];
        out[gi] = s3;
    }
}

// ---------------------------------------------------------------- launch
extern "C" void kernel_launch(void* const* d_in, const int* in_sizes, int n_in,
                              void* d_out, int out_size, void* d_ws, size_t ws_size,
                              hipStream_t stream) {
    const int*   x        = (const int*)d_in[0];
    const int*   eidx     = (const int*)d_in[1];   // [2, N_EDGES]: row0=src, row1=dst
    const int*   batch    = (const int*)d_in[3];
    const float* node_emb = (const float*)d_in[4];
    const float* cW1      = (const float*)d_in[6];
    const float* cb1      = (const float*)d_in[7];
    const float* cW2      = (const float*)d_in[8];
    const float* cb2      = (const float*)d_in[9];
    const float* mW1      = (const float*)d_in[10];
    const float* mb1      = (const float*)d_in[11];
    const float* mW2      = (const float*)d_in[12];
    const float* mb2      = (const float*)d_in[13];
    const float* mW3      = (const float*)d_in[14];
    const float* mb3      = (const float*)d_in[15];
    // edge_attr (d_in[2]) / edge_emb (d_in[5]) are dead in the reference.

    float* h   = (float*)d_ws;                       // 51.2 MB
    float* agg = h + (size_t)N_NODES * HID;          // 51.2 MB
    float* g   = agg + (size_t)N_NODES * HID;        // 256 KB
    float* out = (float*)d_out;

    const size_t hbytes = (size_t)N_NODES * HID * sizeof(float);
    const int nh_blocks = (N_NODES * HID + 255) / 256;    // 50000
    const int ne_blocks = (N_EDGES * HID + 255) / 256;    // 312500
    const int gm_blocks = (N_NODES + 63) / 64;            // 1563

    embed_kernel<<<nh_blocks, 256, 0, stream>>>(x, node_emb, h);

    for (int l = 0; l < N_LAYERS; ++l) {
        // agg starts as h, scatter adds h[src] on top -> agg holds (sum + h) = z-input
        hipMemcpyAsync(agg, h, hbytes, hipMemcpyDeviceToDevice, stream);
        scatter_add_kernel<<<ne_blocks, 256, 0, stream>>>(h, eidx, eidx + N_EDGES, agg);
        gine_layer_gemm<<<gm_blocks, 256, 0, stream>>>(
            agg, cW1 + (size_t)l * HID * HID, cb1 + (size_t)l * HID,
                 cW2 + (size_t)l * HID * HID, cb2 + (size_t)l * HID, h);
    }

    zero_kernel<<<(NUM_GRAPHS * HID + 255) / 256, 256, 0, stream>>>(g, NUM_GRAPHS * HID);
    pool_add_kernel<<<nh_blocks, 256, 0, stream>>>(h, batch, g);
    mlp_head_kernel<<<NUM_GRAPHS, 64, 0, stream>>>(g, mW1, mb1, mW2, mb2, mW3, mb3, out);
}
